// SeemoRe3D_43215960932750
// MI455X (gfx1250) — matrix operand order repaired and checked by reading, not compiled, not run on hardware
//
#include <hip/hip_runtime.h>
#include <hip/hip_bf16.h>
#include <math.h>

// ---------------------------------------------------------------------------
// SeemoRe3D forward for MI455X (gfx1250, wave32).
// - 3x3x3 64->64 convs (compute-bound): implicit GEMM on
//   v_wmma_f32_16x16x32_bf16 with double-buffered async-LDS tile staging
//   (global_load_async_to_lds_b128 + s_wait_asynccnt) from a bf16
//   voxel-major mirror of the input.
// - 1x1x1 channel GEMMs (bandwidth-bound): WMMA with sync LDS staging.
// - Depthwise / LN / MoE / resize: coalesced fp32 VALU kernels.
// ---------------------------------------------------------------------------

#define PVOL (64 * 64 * 64) // 262144 voxels

typedef __attribute__((ext_vector_type(16))) __bf16 bf16x16;
typedef __attribute__((ext_vector_type(8)))  __bf16 bf16x8;
typedef __attribute__((ext_vector_type(8)))  float  f32x8;
typedef __attribute__((ext_vector_type(4)))  float  f32x4;

__device__ __forceinline__ float gelu_erf(float x) {
  return 0.5f * x * (1.0f + erff(x * 0.70710678118654752f));
}

__device__ __forceinline__ __bf16 f2bf(float f) {
  unsigned u = __builtin_bit_cast(unsigned, f);
  unsigned r = u + 0x7FFFu + ((u >> 16) & 1u); // round-to-nearest-even
  unsigned short h = (unsigned short)(r >> 16);
  return __builtin_bit_cast(__bf16, h);
}

__device__ __forceinline__ f32x8 zero8() {
  f32x8 z;
#pragma unroll
  for (int e = 0; e < 8; ++e) z[e] = 0.0f;
  return z;
}

// Async 16B global->LDS copy (ASYNCcnt-tracked). lds_byte = LDS byte address
// (low 32 bits of the generic shared-pointer address).
__device__ __forceinline__ void cp_async16(unsigned lds_byte, const void* g) {
  asm volatile("global_load_async_to_lds_b128 %0, %1, off"
               :: "v"(lds_byte), "v"(g) : "memory");
}
#define WAIT_ASYNC(n) asm volatile("s_wait_asynccnt " #n ::: "memory")

__device__ __forceinline__ unsigned lds_addr(const void* p) {
  return (unsigned)(unsigned long long)p;
}

// A-matrix (16x32 bf16, MxK) per-lane pack: two contiguous 16B chunks at
// +8*hi and +16+8*hi within the 32-wide K slice (ISA 16-bit A layout).
__device__ __forceinline__ bf16x16 pack_a(const __bf16* row, int hi) {
  bf16x8 lo = *(const bf16x8*)(row + 8 * hi);
  bf16x8 hh = *(const bf16x8*)(row + 16 + 8 * hi);
  bf16x16 r;
#pragma unroll
  for (int e = 0; e < 8; ++e) { r[e] = lo[e]; r[e + 8] = hh[e]; }
  return r;
}

// B-matrix (32x16 bf16, KxN) per-lane pack: two contiguous 16B chunks at
// K = 8*hi + {0..7} and 16 + 8*hi + {0..7} for column n = lane&15.
__device__ __forceinline__ bf16x16 pack_b(const __bf16* row, int hi) {
  bf16x8 lo = *(const bf16x8*)(row + 8 * hi);
  bf16x8 hh = *(const bf16x8*)(row + 16 + 8 * hi);
  bf16x16 r;
#pragma unroll
  for (int e = 0; e < 8; ++e) { r[e] = lo[e]; r[e + 8] = hh[e]; }
  return r;
}

// ---------------------------------------------------------------------------
// 1x1x1 conv as GEMM: Y[oc,p] = act(sum_ic W[oc,ic]*X[ic,p] + b[oc]) (+Res)
// ---------------------------------------------------------------------------
__global__ __launch_bounds__(256) void k_gemm64(
    const float* __restrict__ X, const float* __restrict__ W,
    const float* __restrict__ Bias, const float* __restrict__ Res,
    float* __restrict__ Y, int P, int act, int shuf) {
  __shared__ __align__(16) __bf16 sX[128 * 64]; // [j][ic]
  __shared__ __align__(16) __bf16 sW[64 * 64];  // [oc][ic]
  __shared__ __align__(16) float  sY[64 * 128]; // [oc][j]
  const int tid = threadIdx.x;
  const int p0 = blockIdx.x * 128;
  const int ocb = blockIdx.y * 64;

  for (int t = tid; t < 64 * 64; t += 256) {
    int oc = t >> 6, ic = t & 63;
    sW[t] = f2bf(W[(ocb + oc) * 64 + ic]);
  }
  for (int t = tid; t < 128 * 64; t += 256) {
    int j = t & 127, ic = t >> 7;
    int p = p0 + j;
    float v = (p < P) ? X[(size_t)ic * P + p] : 0.0f;
    sX[j * 64 + ic] = f2bf(v);
  }
  __syncthreads();

  const int wv = tid >> 5, ln = tid & 31;
  const int m = ln & 15, hi = ln >> 4;
  f32x8 acc[4];
#pragma unroll
  for (int nt = 0; nt < 4; ++nt) acc[nt] = zero8();

  const __bf16* arow = &sX[(wv * 16 + m) * 64];
#pragma unroll
  for (int kk = 0; kk < 2; ++kk) {
    bf16x16 a = pack_a(arow + kk * 32, hi);
#pragma unroll
    for (int nt = 0; nt < 4; ++nt) {
      bf16x16 b = pack_b(&sW[(nt * 16 + m) * 64 + kk * 32], hi);
      acc[nt] = __builtin_amdgcn_wmma_f32_16x16x32_bf16(
          false, a, false, b, (short)0, acc[nt], false, false);
    }
  }
#pragma unroll
  for (int nt = 0; nt < 4; ++nt)
#pragma unroll
    for (int r = 0; r < 8; ++r)
      sY[(nt * 16 + m) * 128 + wv * 16 + (r + 8 * hi)] = acc[nt][r];
  __syncthreads();

  for (int t = tid; t < 64 * 128; t += 256) {
    int oc = t >> 7, j = t & 127;
    int p = p0 + j;
    if (p >= P) continue;
    float v = sY[t] + Bias[ocb + oc];
    if (act) v = gelu_erf(v);
    int mch = ocb + oc;
    if (shuf) mch = 2 * (mch & 63) + (mch >> 6); // channel_shuffle3d groups=2
    size_t o = (size_t)mch * P + p;
    if (Res) v += Res[o];
    Y[o] = v;
  }
}

// ---------------------------------------------------------------------------
// bf16 pre-pack kernels for the async conv3 path.
// ---------------------------------------------------------------------------
// [64][P] f32 -> voxel-major [P][64] bf16 (each voxel row = contiguous 128B)
__global__ __launch_bounds__(256) void k_to_vox(const float* __restrict__ X,
                                                __bf16* __restrict__ Y) {
  int p = blockIdx.x * 256 + threadIdx.x;
  unsigned pk[32];
#pragma unroll
  for (int c = 0; c < 64; c += 2) {
    unsigned lo = (unsigned)__builtin_bit_cast(unsigned short,
                                               f2bf(X[(size_t)c * PVOL + p]));
    unsigned hh = (unsigned)__builtin_bit_cast(unsigned short,
                                               f2bf(X[(size_t)(c + 1) * PVOL + p]));
    pk[c >> 1] = lo | (hh << 16);
  }
  uint4* dst = (uint4*)(Y + (size_t)p * 64);
#pragma unroll
  for (int i = 0; i < 8; ++i)
    dst[i] = make_uint4(pk[4 * i], pk[4 * i + 1], pk[4 * i + 2], pk[4 * i + 3]);
}

// (oc,ic,27) f32 -> [tap][oc][ic] bf16
__global__ __launch_bounds__(256) void k_w3(const float* __restrict__ W,
                                            __bf16* __restrict__ Y) {
  int id = blockIdx.x * 256 + threadIdx.x;
  if (id >= 27 * 64 * 64) return;
  int tap = id >> 12;
  int oc = (id >> 6) & 63, ic = id & 63;
  Y[id] = f2bf(W[(oc * 64 + ic) * 27 + tap]);
}

// ---------------------------------------------------------------------------
// 3x3x3 conv 64->64, implicit GEMM, double-buffered async-LDS staging.
// Per wave per tap: 6 async b128 instructions (4 X-chunk, 2 W-chunk).
// ---------------------------------------------------------------------------
__global__ __launch_bounds__(256) void k_conv3_wmma(
    const __bf16* __restrict__ Xbf, const __bf16* __restrict__ Wbf,
    const float* __restrict__ Bias, float* __restrict__ Y, int act) {
  __shared__ __align__(16) unsigned char smem[2 * 16384 + 2 * 8192]; // 48KB
  __bf16* sX = (__bf16*)smem;            // [2][128*64]
  __bf16* sW = (__bf16*)(smem + 32768);  // [2][64*64]
  float*  sY = (float*)smem;             // epilogue reuse (32KB)

  const int tid = threadIdx.x;
  const int p0 = blockIdx.x * 128;
  const int wv = tid >> 5, ln = tid & 31;
  const int m = ln & 15, hi = ln >> 4;

  f32x8 acc[4];
#pragma unroll
  for (int nt = 0; nt < 4; ++nt) acc[nt] = zero8();

  // issue async copies for a tap's tiles into buffer bb
  auto stage = [&](int tap, int bb) {
    const int dz = tap / 9 - 1, dy = (tap / 3) % 3 - 1, dx = tap % 3 - 1;
    const int off = dz * 4096 + dy * 64 + dx;
#pragma unroll
    for (int i = 0; i < 4; ++i) {
      int ch = tid + i * 256;       // 0..1023 (16B chunks of the X tile)
      int j = ch >> 3, c8 = ch & 7; // voxel row, chunk within row
      int p = p0 + j;
      int x = p & 63, y = (p >> 6) & 63, z = p >> 12;
      bool ok = (unsigned)(x + dx) < 64u && (unsigned)(y + dy) < 64u &&
                (unsigned)(z + dz) < 64u;
      int q = ok ? p + off : p; // clamp to a safe in-range address
      cp_async16(lds_addr(sX + bb * 8192 + ch * 8),
                 Xbf + (size_t)q * 64 + c8 * 8);
    }
#pragma unroll
    for (int i = 0; i < 2; ++i) {
      int ch = tid + i * 256; // 0..511 (16B chunks of the 64x64 W tap)
      cp_async16(lds_addr(sW + bb * 4096 + ch * 8),
                 Wbf + (size_t)tap * 4096 + ch * 8);
    }
  };
  // zero the out-of-volume boundary rows of buffer bb for tap
  auto patch = [&](int tap, int bb) {
    const int dz = tap / 9 - 1, dy = (tap / 3) % 3 - 1, dx = tap % 3 - 1;
#pragma unroll
    for (int i = 0; i < 4; ++i) {
      int ch = tid + i * 256;
      int j = ch >> 3;
      int p = p0 + j;
      int x = p & 63, y = (p >> 6) & 63, z = p >> 12;
      bool ok = (unsigned)(x + dx) < 64u && (unsigned)(y + dy) < 64u &&
                (unsigned)(z + dz) < 64u;
      if (!ok) {
        f32x4 zz;
        zz[0] = 0.f; zz[1] = 0.f; zz[2] = 0.f; zz[3] = 0.f;
        *(f32x4*)(sX + bb * 8192 + ch * 8) = zz;
      }
    }
  };

  stage(0, 0);
  for (int tap = 0; tap < 27; ++tap) {
    const int bb = tap & 1;
    if (tap + 1 < 27) {
      stage(tap + 1, 1 - bb);
      WAIT_ASYNC(6); // current tap's 6 copies landed; next tap in flight
    } else {
      WAIT_ASYNC(0);
    }
    patch(tap, bb);
    __syncthreads();

    const __bf16* arow = &sX[bb * 8192 + (wv * 16 + m) * 64];
#pragma unroll
    for (int kk = 0; kk < 2; ++kk) {
      bf16x16 a = pack_a(arow + kk * 32, hi);
#pragma unroll
      for (int nt = 0; nt < 4; ++nt) {
        bf16x16 b = pack_b(&sW[bb * 4096 + (nt * 16 + m) * 64 + kk * 32], hi);
        acc[nt] = __builtin_amdgcn_wmma_f32_16x16x32_bf16(
            false, a, false, b, (short)0, acc[nt], false, false);
      }
    }
    __syncthreads(); // all waves done reading buf bb before it is re-staged
  }

#pragma unroll
  for (int nt = 0; nt < 4; ++nt)
#pragma unroll
    for (int r = 0; r < 8; ++r)
      sY[(nt * 16 + m) * 128 + wv * 16 + (r + 8 * hi)] = acc[nt][r];
  __syncthreads();
  for (int t = tid; t < 64 * 128; t += 256) {
    int oc = t >> 7, j = t & 127;
    float v = sY[t] + Bias[oc];
    if (act) v = gelu_erf(v);
    Y[(size_t)oc * PVOL + p0 + j] = v;
  }
}

// --------------------------- conv1: 1 -> 64, 3^3 ---------------------------
__global__ __launch_bounds__(256) void k_conv1(
    const float* __restrict__ X, const float* __restrict__ W,
    const float* __restrict__ Bias, float* __restrict__ Y) {
  __shared__ float sw[64 * 27];
  __shared__ float sb[64];
  const int tid = threadIdx.x;
  for (int t = tid; t < 64 * 27; t += 256) sw[t] = W[t];
  if (tid < 64) sb[tid] = Bias[tid];
  __syncthreads();
  int p = blockIdx.x * 256 + tid;
  int x = p & 63, y = (p >> 6) & 63, z = p >> 12;
  float nb[27];
#pragma unroll
  for (int t = 0; t < 27; ++t) {
    int dz = t / 9 - 1, dy = (t / 3) % 3 - 1, dx = t % 3 - 1;
    bool ok = (unsigned)(x + dx) < 64u && (unsigned)(y + dy) < 64u &&
              (unsigned)(z + dz) < 64u;
    nb[t] = ok ? X[p + dz * 4096 + dy * 64 + dx] : 0.0f;
  }
  for (int oc = 0; oc < 64; ++oc) {
    float a = sb[oc];
#pragma unroll
    for (int t = 0; t < 27; ++t) a += sw[oc * 27 + t] * nb[t];
    Y[(size_t)oc * PVOL + p] = a;
  }
}

// ----------------------- per-voxel channel LayerNorm -----------------------
__global__ __launch_bounds__(256) void k_ln(
    const float* __restrict__ X, const float* __restrict__ w,
    const float* __restrict__ b, float* __restrict__ Y) {
  int p = blockIdx.x * 256 + threadIdx.x;
  float s = 0.0f, s2 = 0.0f;
  for (int c = 0; c < 64; ++c) {
    float v = X[(size_t)c * PVOL + p];
    s += v; s2 += v * v;
  }
  float mu = s * (1.0f / 64.0f);
  float var = s2 * (1.0f / 64.0f) - mu * mu;
  float inv = rsqrtf(var + 1e-6f);
  for (int c = 0; c < 64; ++c)
    Y[(size_t)c * PVOL + p] = (X[(size_t)c * PVOL + p] - mu) * inv * w[c] + b[c];
}

// ---------------- depthwise 1-D conv along axis (k = 3 or 11) --------------
__global__ __launch_bounds__(256) void k_dwaxis(
    const float* __restrict__ X, const float* __restrict__ W,
    const float* __restrict__ B, float* __restrict__ Y,
    int axis, int K, int act) {
  int id = blockIdx.x * 256 + threadIdx.x;
  int p = id & (PVOL - 1);
  int c = id >> 18;
  int x = p & 63, y = (p >> 6) & 63, z = p >> 12;
  int coord = (axis == 0) ? x : (axis == 1) ? y : z;
  int stride = (axis == 0) ? 1 : (axis == 1) ? 64 : 4096;
  int h = K >> 1;
  float a = B[c];
  for (int t = 0; t < K; ++t) {
    int cc = coord + t - h;
    if ((unsigned)cc < 64u)
      a += W[c * K + t] * X[(size_t)c * PVOL + p + (t - h) * stride];
  }
  if (act) a = gelu_erf(a);
  Y[(size_t)c * PVOL + p] = a;
}

// ------------------- depthwise 3^3 conv on DIM^3 (DIM=2^LOG) ----------------
__global__ __launch_bounds__(256) void k_dw3(
    const float* __restrict__ X, const float* __restrict__ W,
    const float* __restrict__ B, float* __restrict__ Y, int LOG, int act) {
  int DIM = 1 << LOG;
  int Pv = 1 << (3 * LOG);
  int id = blockIdx.x * 256 + threadIdx.x;
  if (id >= 64 * Pv) return;
  int p = id & (Pv - 1);
  int c = id >> (3 * LOG);
  int x = p & (DIM - 1), y = (p >> LOG) & (DIM - 1), z = p >> (2 * LOG);
  float a = B[c];
  for (int t = 0; t < 27; ++t) {
    int dz = t / 9 - 1, dy = (t / 3) % 3 - 1, dx = t % 3 - 1;
    int sx = x + dx, sy = y + dy, sz = z + dz;
    if ((unsigned)sx < (unsigned)DIM && (unsigned)sy < (unsigned)DIM &&
        (unsigned)sz < (unsigned)DIM)
      a += W[c * 27 + t] * X[(size_t)c * Pv + (((sz << LOG) + sy) << LOG) + sx];
  }
  if (act) a = gelu_erf(a);
  Y[(size_t)c * Pv + p] = a;
}

// ------------- depthwise 4^3 stride-4 (no pad) + GELU: DI^3 -> (DI/4)^3 -----
__global__ __launch_bounds__(256) void k_agg(
    const float* __restrict__ X, const float* __restrict__ W,
    const float* __restrict__ B, float* __restrict__ Y, int LOGI) {
  int LOGO = LOGI - 2;
  int Pi = 1 << (3 * LOGI), Po = 1 << (3 * LOGO), DO = 1 << LOGO;
  int id = blockIdx.x * 256 + threadIdx.x;
  if (id >= 64 * Po) return;
  int p = id & (Po - 1);
  int c = id >> (3 * LOGO);
  int ox = p & (DO - 1), oy = (p >> LOGO) & (DO - 1), oz = p >> (2 * LOGO);
  float a = B[c];
  for (int tz = 0; tz < 4; ++tz)
    for (int ty = 0; ty < 4; ++ty)
      for (int tx = 0; tx < 4; ++tx) {
        int ix = ox * 4 + tx, iy = oy * 4 + ty, iz = oz * 4 + tz;
        a += W[c * 64 + (tz * 4 + ty) * 4 + tx] *
             X[(size_t)c * Pi + (iz << (2 * LOGI)) + (iy << LOGI) + ix];
      }
  Y[(size_t)c * Po + p] = gelu_erf(a);
}

// ----------- trilinear upsample 4^3 -> 64^3 (half-pixel) fused add ----------
__global__ __launch_bounds__(256) void k_up_add(
    const float* __restrict__ X, float* __restrict__ Y) {
  int id = blockIdx.x * 256 + threadIdx.x;
  int p = id & (PVOL - 1);
  int c = id >> 18;
  int x = p & 63, y = (p >> 6) & 63, z = p >> 12;
  float fx = (x + 0.5f) * 0.0625f - 0.5f;
  float fy = (y + 0.5f) * 0.0625f - 0.5f;
  float fz = (z + 0.5f) * 0.0625f - 0.5f;
  int x0 = (int)floorf(fx), y0 = (int)floorf(fy), z0 = (int)floorf(fz);
  float tx = fx - x0, ty = fy - y0, tz = fz - z0;
  int x0c = min(max(x0, 0), 3), x1c = min(max(x0 + 1, 0), 3);
  int y0c = min(max(y0, 0), 3), y1c = min(max(y0 + 1, 0), 3);
  int z0c = min(max(z0, 0), 3), z1c = min(max(z0 + 1, 0), 3);
  const float* Xc = X + (size_t)c * 64;
#define SAMP(zz, yy, xx) Xc[((zz) * 4 + (yy)) * 4 + (xx)]
  float c00 = SAMP(z0c, y0c, x0c) * (1 - tx) + SAMP(z0c, y0c, x1c) * tx;
  float c01 = SAMP(z0c, y1c, x0c) * (1 - tx) + SAMP(z0c, y1c, x1c) * tx;
  float c10 = SAMP(z1c, y0c, x0c) * (1 - tx) + SAMP(z1c, y0c, x1c) * tx;
  float c11 = SAMP(z1c, y1c, x0c) * (1 - tx) + SAMP(z1c, y1c, x1c) * tx;
#undef SAMP
  float v = (c00 * (1 - ty) + c01 * ty) * (1 - tz) +
            (c10 * (1 - ty) + c11 * ty) * tz;
  Y[(size_t)c * PVOL + p] += v;
}

// --------------------------- elementwise binop ------------------------------
__global__ __launch_bounds__(256) void k_binop(
    const float* __restrict__ A, const float* __restrict__ B,
    float* __restrict__ O, int n, int op) {
  int id = blockIdx.x * 256 + threadIdx.x;
  if (id < n) O[id] = op ? A[id] * B[id] : A[id] + B[id];
}

// ------------------------ spatial mean pool per channel ---------------------
__global__ __launch_bounds__(256) void k_pool(
    const float* __restrict__ X, float* __restrict__ pooled) {
  __shared__ float red[256];
  int c = blockIdx.x;
  float s = 0.0f;
  for (int p = threadIdx.x; p < PVOL; p += 256) s += X[(size_t)c * PVOL + p];
  red[threadIdx.x] = s;
  __syncthreads();
  for (int k = 128; k > 0; k >>= 1) {
    if (threadIdx.x < k) red[threadIdx.x] += red[threadIdx.x + k];
    __syncthreads();
  }
  if (threadIdx.x == 0) pooled[c] = red[0] * (1.0f / PVOL);
}

// ------------------------ gate: softmax + top-1 -----------------------------
__global__ void k_gate(const float* __restrict__ pooled,
                       const float* __restrict__ gate, float* __restrict__ sel) {
  if (threadIdx.x == 0) {
    float l[3];
    for (int i = 0; i < 3; ++i) {
      float s = 0.0f;
      for (int c = 0; c < 64; ++c) s += pooled[c] * gate[i * 64 + c];
      l[i] = s;
    }
    float mx = fmaxf(l[0], fmaxf(l[1], l[2]));
    float e0 = expf(l[0] - mx), e1 = expf(l[1] - mx), e2 = expf(l[2] - mx);
    float sum = e0 + e1 + e2;
    int bi = 0; float bl = l[0];
    if (l[1] > bl) { bl = l[1]; bi = 1; }
    if (l[2] > bl) { bl = l[2]; bi = 2; }
    sel[0] = expf(bl - mx) / sum;
    sel[1] = (float)bi;
  }
}

// ------------- fused top-1 MoE expert: out = xh + w* C3((C1 xh)*(C2 k)) -----
__global__ __launch_bounds__(256) void k_expert(
    const float* __restrict__ xh, const float* __restrict__ kb,
    const float* __restrict__ sel,
    const float* c1w0, const float* c1b0, const float* c2w0, const float* c2b0,
    const float* c3w0, const float* c3b0,
    const float* c1w1, const float* c1b1, const float* c2w1, const float* c2b1,
    const float* c3w1, const float* c3b1,
    const float* c1w2, const float* c1b2, const float* c2w2, const float* c2b2,
    const float* c3w2, const float* c3b2,
    float* __restrict__ O) {
  int p = blockIdx.x * 256 + threadIdx.x;
  int ei = (int)sel[1];
  float wsr = sel[0];
  const float* c1w = (ei == 0) ? c1w0 : (ei == 1) ? c1w1 : c1w2;
  const float* c1b = (ei == 0) ? c1b0 : (ei == 1) ? c1b1 : c1b2;
  const float* c2w = (ei == 0) ? c2w0 : (ei == 1) ? c2w1 : c2w2;
  const float* c2b = (ei == 0) ? c2b0 : (ei == 1) ? c2b1 : c2b2;
  const float* c3w = (ei == 0) ? c3w0 : (ei == 1) ? c3w1 : c3w2;
  const float* c3b = (ei == 0) ? c3b0 : (ei == 1) ? c3b1 : c3b2;
  int ld = 2 << ei; // 2, 4, 8
  float u[8], v[8];
  for (int l = 0; l < ld; ++l) { u[l] = c1b[l]; v[l] = c2b[l]; }
  for (int ic = 0; ic < 64; ++ic) {
    float xv = xh[(size_t)ic * PVOL + p];
    float kv = kb[(size_t)ic * PVOL + p];
    for (int l = 0; l < ld; ++l) {
      u[l] += c1w[l * 64 + ic] * xv;
      v[l] += c2w[l * 64 + ic] * kv;
    }
  }
  float g[8];
  for (int l = 0; l < ld; ++l) g[l] = u[l] * v[l];
  for (int oc = 0; oc < 64; ++oc) {
    float s = c3b[oc];
    for (int l = 0; l < ld; ++l) s += c3w[oc * ld + l] * g[l];
    O[(size_t)oc * PVOL + p] = xh[(size_t)oc * PVOL + p] + wsr * s;
  }
}

// ---------------------------------------------------------------------------
// Driver. Input leaves follow JAX pytree order (dict keys sorted):
//  0 conv1.b 1 conv1.w 2 conv2.b 3 conv2.w
//  per layer l: rme base R = 4 + 57*l (37 leaves), sme base S = R+37 (20)
//  118 normb 119 normw 120 x
// ---------------------------------------------------------------------------
extern "C" void kernel_launch(void* const* d_in, const int* in_sizes, int n_in,
                              void* d_out, int out_size, void* d_ws, size_t ws_size,
                              hipStream_t stream) {
  (void)in_sizes; (void)n_in; (void)out_size; (void)ws_size;
  const int P = PVOL;
  auto F = [&](int i) { return (const float*)d_in[i]; };

  float* w = (float*)d_ws;
  float* X    = w;
  float* A1   = w + (size_t)64 * P;
  float* A2   = w + (size_t)128 * P;
  float* B128 = w + (size_t)192 * P;      // 128 channels
  float* H0   = B128;                     // first 64 rows
  float* H1   = B128 + (size_t)64 * P;    // second 64 rows
  float* SM   = w + (size_t)320 * P;
  float* pooled = SM;
  float* sel    = SM + 64;
  float* pyr16  = SM + 128;               // 64 * 16^3
  float* pyr4a  = pyr16 + 64 * 4096;      // 64 * 4^3
  float* pyr4b  = pyr4a + 64 * 64;
  float* pyr4c  = pyr4b + 64 * 64;
  __bf16* Wbf   = (__bf16*)(SM + 274560);          // 27*64*64 bf16
  __bf16* Xbf   = (__bf16*)(SM + 274560 + 55296);  // P*64 bf16 voxel-major

  dim3 blk(256);
  dim3 gP(2048, 1), gP2(2048, 2);

  // x -> conv1
  k_conv1<<<1024, blk, 0, stream>>>(F(120), F(1), F(0), X);

  for (int l = 0; l < 2; ++l) {
    int R = 4 + 57 * l;
    int S = R + 37;
    // ---- RME ----
    // gelu(conv3 c1a) via async WMMA conv
    k_w3<<<432, blk, 0, stream>>>(F(R + 3), Wbf);
    k_to_vox<<<1024, blk, 0, stream>>>(X, Xbf);
    k_conv3_wmma<<<gP, blk, 0, stream>>>(Xbf, Wbf, F(R + 2), A1, 1);
    // c1b 64->128 + channel shuffle (xh = H0, k = H1)
    k_gemm64<<<gP2, blk, 0, stream>>>(A1, F(R + 5), F(R + 4), nullptr, B128, P, 0, 1);
    // striped sconv on xh (W,H,D), gelu on last pass -> A1
    k_dwaxis<<<65536, blk, 0, stream>>>(H0, F(R + 32), F(R + 31), A1, 0, 3, 0);
    k_dwaxis<<<65536, blk, 0, stream>>>(A1, F(R + 34), F(R + 33), A2, 1, 3, 0);
    k_dwaxis<<<65536, blk, 0, stream>>>(A2, F(R + 36), F(R + 35), A1, 2, 3, 1);
    // calibrate(k): agg x2 -> cal_dw -> cal_pw -> trilinear up, k += up
    k_agg<<<1024, blk, 0, stream>>>(H1, F(R + 1), F(R + 0), pyr16, 6);
    k_agg<<<16, blk, 0, stream>>>(pyr16, F(R + 1), F(R + 0), pyr4a, 4);
    k_dw3<<<16, blk, 0, stream>>>(pyr4a, F(R + 7), F(R + 6), pyr4b, 2, 0);
    k_gemm64<<<dim3(1, 1), blk, 0, stream>>>(pyr4b, F(R + 9), F(R + 8), nullptr,
                                             pyr4c, 64, 0, 0);
    k_up_add<<<65536, blk, 0, stream>>>(pyr4c, H1);
    // MoE: pool -> gate -> top-1 expert; out = xh + w* eo -> A2
    k_pool<<<64, blk, 0, stream>>>(A1, pooled);
    k_gate<<<1, 32, 0, stream>>>(pooled, F(R + 28), sel);
    k_expert<<<1024, blk, 0, stream>>>(A1, H1, sel,
        F(R + 11), F(R + 10), F(R + 13), F(R + 12), F(R + 15), F(R + 14),
        F(R + 17), F(R + 16), F(R + 19), F(R + 18), F(R + 21), F(R + 20),
        F(R + 23), F(R + 22), F(R + 25), F(R + 24), F(R + 27), F(R + 26),
        A2);
    // proj -> X
    k_gemm64<<<gP, blk, 0, stream>>>(A2, F(R + 30), F(R + 29), nullptr, X, P, 0, 0);

    // ---- SME ----
    k_ln<<<1024, blk, 0, stream>>>(X, F(S + 13), F(S + 12), A1);           // n1
    k_gemm64<<<gP2, blk, 0, stream>>>(A1, F(S + 19), F(S + 18), nullptr,
                                      B128, P, 1, 0);                       // gelu(to_qv)
    k_dwaxis<<<65536, blk, 0, stream>>>(H0, F(S + 1), F(S + 0), A1, 0, 11, 0);
    k_dwaxis<<<65536, blk, 0, stream>>>(A1, F(S + 3), F(S + 2), A2, 1, 11, 0);
    k_dwaxis<<<65536, blk, 0, stream>>>(A2, F(S + 5), F(S + 4), A1, 2, 11, 0);
    k_binop<<<65536, blk, 0, stream>>>(A1, H1, A2, 64 * P, 1);              // q*v
    k_gemm64<<<gP, blk, 0, stream>>>(A2, F(S + 17), F(S + 16), X, X, P, 0, 0); // proj + x
    k_ln<<<1024, blk, 0, stream>>>(X, F(S + 15), F(S + 14), A1);           // n2
    k_gemm64<<<gP2, blk, 0, stream>>>(A1, F(S + 9), F(S + 8), nullptr,
                                      B128, P, 1, 0);                       // gelu(fn1)
    k_dw3<<<65536, blk, 0, stream>>>(H1, F(S + 7), F(S + 6), A1, 6, 0);     // ffn_gate(g2)
    k_binop<<<65536, blk, 0, stream>>>(H0, A1, A2, 64 * P, 1);              // a*g2
    k_gemm64<<<gP, blk, 0, stream>>>(A2, F(S + 11), F(S + 10), X, X, P, 1, 0); // gelu(fn2)+x
  }

  // final norm + conv2 -> d_out
  k_ln<<<1024, blk, 0, stream>>>(X, F(119), F(118), A1);
  k_w3<<<432, blk, 0, stream>>>(F(3), Wbf);
  k_to_vox<<<1024, blk, 0, stream>>>(A1, Xbf);
  k_conv3_wmma<<<gP, blk, 0, stream>>>(Xbf, Wbf, F(2), (float*)d_out, 0);
}
